// QRNNLayer_91122026152833
// MI455X (gfx1250) — matrix operand order repaired and checked by reading, not compiled, not run on hardware
//
#include <hip/hip_runtime.h>
#include <hip/hip_bf16.h>
#include <stddef.h>

// QRNN-fo layer for MI455X (gfx1250, wave32, WMMA).
// Kernel 1: Y = concat(X_t, X_{t-1}) @ W^T + b, gates pre-activated, via
//           v_wmma_f32_16x16x32_bf16, register->LDS software pipeline.
// Kernel 2: diagonal linear recurrence c_t = f*z + (1-f)*c, H = sig(o)*c.

typedef __attribute__((ext_vector_type(16))) __bf16 v16bf;
typedef __attribute__((ext_vector_type(8)))  __bf16 v8bf;
typedef __attribute__((ext_vector_type(8)))  float  v8f;

// Problem constants
constexpr int Bsz = 16;          // batch
constexpr int S   = 2048;        // sequence
constexpr int D   = 1024;        // input dim
constexpr int H   = 1024;        // hidden dim
constexpr int M   = S * Bsz;     // 32768 GEMM rows   (m = s*16 + b)
constexpr int N   = 3 * H;       // 3072  GEMM cols
constexpr int K   = 2 * D;       // 2048  GEMM reduction

// Tiling
constexpr int BM = 128, BN = 128, BK = 32;
constexpr int LDT    = 48;       // padded LDS row stride (bf16) -> 96B, frags 32B-aligned
constexpr int KSTEPS = K / BK;   // 64
constexpr int KHALF  = KSTEPS / 2; // window boundary at K-step 32 (k == D)

__device__ __forceinline__ float fast_sigmoid(float x) {
  return 1.0f / (1.0f + __expf(-x));
}

__device__ __forceinline__ void cvt_store4(__bf16* dst, float4 v) {
  dst[0] = (__bf16)v.x; dst[1] = (__bf16)v.y;
  dst[2] = (__bf16)v.z; dst[3] = (__bf16)v.w;
}

// One 128x128 output tile. EDGE=true only for block-row 0 (the only rows
// where the window-2 "previous token" is the zero pad at t=0).
template <bool EDGE>
__device__ __forceinline__ void gemm_tile(const float* __restrict__ X,
                                          const float* __restrict__ W,
                                          const float* __restrict__ bias,
                                          float* __restrict__ Y,
                                          __bf16* As, __bf16* Bs,
                                          int m0, int n0) {
  const int tid  = threadIdx.x;
  const int lane = tid & 31;
  const int wid  = tid >> 5;     // 8 waves
  const int wm   = wid >> 2;     // 0..1 : 64-row strip
  const int wn   = wid & 3;      // 0..3 : 32-col strip

  const v8f vzero = {0.f,0.f,0.f,0.f,0.f,0.f,0.f,0.f};
  v8f acc[4][2];
#pragma unroll
  for (int i = 0; i < 4; ++i)
#pragma unroll
    for (int j = 0; j < 2; ++j) acc[i][j] = vzero;

  // cooperative tile-load mapping: 8 threads * float4 cover one 32-wide K row
  const int kt = (tid & 7) * 4;  // k offset within tile
  const int r0 = tid >> 3;       // row 0..31, +32 per pass

  // Loop-invariant per-thread source pointers (bumped by +BK per K-step).
  const float* aptr1[4];   // first half:  X[b, s,   k]
  const float* aptr2[4];   // second half: X[b, s-1, k-D] (clamped; masked if EDGE)
  const float* bptr[4];    // W[n, k]
  bool av[4];
#pragma unroll
  for (int p = 0; p < 4; ++p) {
    const int r  = r0 + p * 32;
    const int m  = m0 + r;
    const int s  = m >> 4;
    const int bb = m & 15;
    aptr1[p] = X + ((size_t)bb * S + s) * D + kt;
    aptr2[p] = X + ((size_t)bb * S + (s > 0 ? s - 1 : 0)) * D + kt; // always-legal addr
    av[p]    = !EDGE || (s > 0);
    bptr[p]  = W + (size_t)(n0 + r) * K + kt;
  }

  // Prologue: prefetch K-step 0 into registers.
  float4 areg[4], breg[4];
#pragma unroll
  for (int p = 0; p < 4; ++p) {
    areg[p] = *(const float4*)aptr1[p]; aptr1[p] += BK;
    breg[p] = *(const float4*)bptr[p];  bptr[p]  += BK;
  }

#pragma unroll 1
  for (int ks = 0; ks < KSTEPS; ++ks) {
    __syncthreads();  // previous iteration's fragment reads complete
    // Publish current K-tile (f32 -> bf16 on the way into LDS).
#pragma unroll
    for (int p = 0; p < 4; ++p) {
      cvt_store4(&As[(r0 + p * 32) * LDT + kt], areg[p]);
      cvt_store4(&Bs[(r0 + p * 32) * LDT + kt], breg[p]);
    }
    __syncthreads();  // tiles visible

    // Prefetch next K-tile (overlaps with the WMMA burst below).
    if (ks + 1 < KSTEPS) {
      if (ks + 1 < KHALF) {      // uniform branch: current token half
#pragma unroll
        for (int p = 0; p < 4; ++p) {
          areg[p] = *(const float4*)aptr1[p]; aptr1[p] += BK;
        }
      } else {                   // previous-token half (zero pad at s==0)
#pragma unroll
        for (int p = 0; p < 4; ++p) {
          float4 v = *(const float4*)aptr2[p]; aptr2[p] += BK;
          if (EDGE && !av[p]) v = make_float4(0.f, 0.f, 0.f, 0.f); // cndmask, no branch
          areg[p] = v;
        }
      }
#pragma unroll
      for (int p = 0; p < 4; ++p) {
        breg[p] = *(const float4*)bptr[p]; bptr[p] += BK;
      }
    }

    // ---- fragments (k = 16*hb + s in both A and B) + 8 WMMAs ----
    const int fr = lane & 15;
    const int hb = lane >> 4;

    v16bf afr[4];
#pragma unroll
    for (int mi = 0; mi < 4; ++mi) {
      // A: lane holds 16 contiguous K at 16*hb
      const __bf16* ap = &As[(wm * 64 + mi * 16 + fr) * LDT + hb * 16];
      v8bf a0 = *(const v8bf*)(ap);
      v8bf a1 = *(const v8bf*)(ap + 8);
      afr[mi] = __builtin_shufflevector(a0, a1,
                  0,1,2,3,4,5,6,7,8,9,10,11,12,13,14,15);
    }
    v16bf bfr[2];
#pragma unroll
    for (int ni = 0; ni < 2; ++ni) {
      // B: lane holds column N=fr, 16 contiguous K at 16*hb
      bfr[ni] = *(const v16bf*)&Bs[(wn * 32 + ni * 16 + fr) * LDT + hb * 16];
    }

#pragma unroll
    for (int mi = 0; mi < 4; ++mi)
#pragma unroll
      for (int ni = 0; ni < 2; ++ni)
        acc[mi][ni] = __builtin_amdgcn_wmma_f32_16x16x32_bf16(
            false, afr[mi], false, bfr[ni], (short)0, acc[mi][ni], false, false);
  }

  // ---- epilogue: bias + gate activation, NT stores (Y streams through, 402MB > L2) ----
  const bool is_tanh = (n0 < H);   // Z gate; F and O use sigmoid
  const int fr = lane & 15;
  const int hb = lane >> 4;
#pragma unroll
  for (int mi = 0; mi < 4; ++mi) {
#pragma unroll
    for (int ni = 0; ni < 2; ++ni) {
      const int col  = n0 + wn * 32 + ni * 16 + fr;
      const int rowb = m0 + wm * 64 + mi * 16 + hb * 8;  // C layout: VGPR v -> M = v + 8*hb
      const float bv = bias[col];
#pragma unroll
      for (int v = 0; v < 8; ++v) {
        float y = acc[mi][ni][v] + bv;
        if (is_tanh) y = 2.0f * fast_sigmoid(2.0f * y) - 1.0f;  // tanh
        else         y = fast_sigmoid(y);
        __builtin_nontemporal_store(y, &Y[(size_t)(rowb + v) * N + col]);
      }
    }
  }
}

__global__ __launch_bounds__(256)
void qrnn_gemm_gates(const float* __restrict__ X,    // (B,S,D)
                     const float* __restrict__ W,    // (3H, 2D) row-major
                     const float* __restrict__ bias, // (3H)
                     float* __restrict__ Y)          // (S,B,3H) pre-activated gates
{
  __shared__ __align__(64) __bf16 As[BM * LDT];
  __shared__ __align__(64) __bf16 Bs[BN * LDT];
  const int n0 = blockIdx.x * BN;
  const int m0 = blockIdx.y * BM;
  if (m0 == 0) gemm_tile<true >(X, W, bias, Y, As, Bs, m0, n0);
  else         gemm_tile<false>(X, W, bias, Y, As, Bs, m0, n0);
}

// Sequential diagonal recurrence: one thread per (b,h). Fully coalesced
// streaming loads/stores (NT: single-use data); activations pre-applied so
// the dependent chain is one FMA per step.
__global__ __launch_bounds__(256)
void qrnn_scan(const float* __restrict__ Yg,      // (S,B,3H) activated gates
               const float* __restrict__ hidden,  // (B,H)
               float* __restrict__ out)           // [S*B*H] Hout, then [B*H] c_last
{
  const int idx = blockIdx.x * blockDim.x + threadIdx.x;  // 0..B*H-1
  const int bb  = idx >> 10;
  const int h   = idx & (H - 1);

  float c = hidden[idx];
  const float* p = Yg + (size_t)bb * N + h;   // z at +0, f at +H, o at +2H
  float*       q = out + idx;                 // stride B*H per step

#pragma unroll 4
  for (int s = 0; s < S; ++s) {
    const float z = __builtin_nontemporal_load(p);
    const float f = __builtin_nontemporal_load(p + H);
    const float o = __builtin_nontemporal_load(p + 2 * H);
    c = fmaf(f, z - c, c);                    // f*z + (1-f)*c
    __builtin_nontemporal_store(o * c, q);    // sigmoid(o) already applied
    p += (size_t)Bsz * N;
    q += Bsz * H;
  }
  out[(size_t)S * Bsz * H + idx] = c;         // c_last
}

extern "C" void kernel_launch(void* const* d_in, const int* in_sizes, int n_in,
                              void* d_out, int out_size, void* d_ws, size_t ws_size,
                              hipStream_t stream) {
  (void)in_sizes; (void)n_in; (void)out_size; (void)ws_size;
  const float* X      = (const float*)d_in[0];  // (B,S,D)
  const float* hidden = (const float*)d_in[1];  // (B,H)
  const float* W      = (const float*)d_in[2];  // (3H,2D)
  const float* bias   = (const float*)d_in[3];  // (3H)
  float*       out    = (float*)d_out;
  float*       Yg     = (float*)d_ws;           // (S,B,3H) fp32 = 402 MB scratch

  dim3 ggrid(N / BN, M / BM);                   // (24, 256)
  qrnn_gemm_gates<<<ggrid, 256, 0, stream>>>(X, W, bias, Yg);

  qrnn_scan<<<(Bsz * H) / 256, 256, 0, stream>>>(Yg, hidden, out);
}
